// MAELossWithKLMessageReg_13675175871043
// MI455X (gfx1250) — compile-verified, matrix operand order checked by reading, not run on hardware
//
#include <hip/hip_runtime.h>
#include <hip/hip_bf16.h>
#include <math.h>

typedef __attribute__((ext_vector_type(16))) __bf16 v16bf;
typedef __attribute__((ext_vector_type(8)))  float  v8f;

#define N_NODES 100000
#define N_EDGES 1000000
#define D_FEAT  6
#define MSG_DIM 100

#define KPAD 320        // hidden dim 300 padded to 10 K-tiles of 32 (col 300 = bias row)
#define NT_H 20         // hidden N tiles (320/16)
#define KT_H 10         // hidden K tiles (320/32)
#define NT_O 13         // output N tiles (208 >= 200)
#define WAVES 4         // waves per block (wave32)

// ---- workspace layout (bytes) ----
#define W1P_CNT (1 * NT_H * 512)        // 10240 bf16
#define W2P_CNT (KT_H * NT_H * 512)     // 102400 bf16
#define W3P_CNT (KT_H * NT_O * 512)     // 66560 bf16
#define W1P_OFF 0
#define W2P_OFF 20480
#define W3P_OFF 225280

__device__ __forceinline__ void lds_fence() {
  asm volatile("s_wait_dscnt 0" ::: "memory");
}

// A operand (16xK bf16 tile) from per-wave LDS h-buffer [16][KPAD], row-major.
// Lane l (l<16): M=l, K = kt*32 + {0..7, 16..23}; lanes 16..31: K += 8.
__device__ __forceinline__ v16bf load_a(const __bf16* hb, int lane, int kt) {
  const int m  = lane & 15;
  const int k0 = kt * 32 + ((lane & 16) >> 1);
  const uint4* p = (const uint4*)(hb + m * KPAD + k0);
  union { uint4 u[2]; v16bf v; } t;
  t.u[0] = p[0];   // K = k0 .. k0+7
  t.u[1] = p[2];   // K = k0+16 .. k0+23
  return t.v;
}

// B operand: pre-swizzled in workspace so each lane reads its 16 bf16 contiguously.
__device__ __forceinline__ v16bf load_b(const __bf16* wp, int tile, int lane) {
  return *(const v16bf*)(wp + (((tile << 5) + lane) << 4));
}

// ---- weight prep: fp32 -> bf16, WMMA B-operand layout, bias folded in as K-row ----
__device__ __forceinline__ void prep_w(int t, int NT, int Kreal, int Nreal,
                                       const float* W, const float* bias,
                                       __bf16* outp) {
  const int e    = t & 15;
  const int lane = (t >> 4) & 31;
  const int tile = t >> 9;
  const int nt   = tile % NT;
  const int kt   = tile / NT;
  const int koff = e + (e & 8) + ((lane & 16) >> 1);
  const int k    = kt * 32 + koff;
  const int n    = nt * 16 + (lane & 15);
  float v = 0.0f;
  if (n < Nreal) {
    if (k < Kreal)       v = W[k * Nreal + n];
    else if (k == Kreal) v = bias[n];      // bias row: pairs with constant-1 activation col
  }
  outp[t] = (__bf16)v;
}

__global__ void prep_kernel(const float* __restrict__ W1, const float* __restrict__ b1,
                            const float* __restrict__ W2, const float* __restrict__ b2,
                            const float* __restrict__ W3, const float* __restrict__ b3,
                            __bf16* w1p, __bf16* w2p, __bf16* w3p) {
  int i = blockIdx.x * blockDim.x + threadIdx.x;
  if (i < W1P_CNT) { prep_w(i, NT_H, 12, 300, W1, b1, w1p); return; }
  i -= W1P_CNT;
  if (i < W2P_CNT) { prep_w(i, NT_H, 300, 300, W2, b2, w2p); return; }
  i -= W2P_CNT;
  if (i < W3P_CNT) { prep_w(i, NT_O, 300, 200, W3, b3, w3p); return; }
}

// ---------------- scalar-output helpers ----------------
__global__ void zero_out_kernel(float* out) { out[0] = 0.0f; }

__global__ void base_loss_kernel(const float* __restrict__ y,
                                 const float* __restrict__ tgt,
                                 float* __restrict__ out) {
  __shared__ float s;
  if (threadIdx.x == 0) s = 0.0f;
  __syncthreads();
  const int i = blockIdx.x * blockDim.x + threadIdx.x;
  float v = 0.0f;
  if (i < N_NODES * 2) v = fabsf(y[i] - tgt[i]);
  atomicAdd(&s, v);
  __syncthreads();
  if (threadIdx.x == 0) atomicAdd(out, s * (1.0f / (float)N_NODES));
}

// ---------------- fused edge-MLP + KL, WMMA bf16 ----------------
__global__ __launch_bounds__(32 * WAVES) void fused_edge_mlp(
    const float* __restrict__ x, const int* __restrict__ ei,
    const __bf16* __restrict__ w1p, const __bf16* __restrict__ w2p,
    const __bf16* __restrict__ w3p, float* __restrict__ out) {
  __shared__ __bf16 hbuf_all[WAVES][16 * KPAD];  // 40 KB
  __shared__ float klsum;
  const int tid  = threadIdx.x;
  const int wave = tid >> 5;
  const int lane = tid & 31;
  __bf16* hb = hbuf_all[wave];
  if (tid == 0) klsum = 0.0f;
  __syncthreads();

  const int ncol  = lane & 15;
  const int mbase = (lane < 16) ? 0 : 8;
  const __bf16 one = (__bf16)1.0f;

  // ---- gather 16 edges: lanes 0..15 fetch src features, 16..31 fetch dst ----
  {
    const int r    = lane & 15;
    const int side = lane >> 4;
    const int e0   = (blockIdx.x * WAVES + wave) * 16;
    const int idx  = ei[side * N_EDGES + (e0 + r)];
    const float* xr = x + idx * D_FEAT;
#pragma unroll
    for (int j = 0; j < D_FEAT; ++j)
      hb[r * KPAD + side * D_FEAT + j] = (__bf16)xr[j];
    uint32_t* z = (uint32_t*)(hb + r * KPAD);
#pragma unroll
    for (int i = 0; i < 5; ++i) z[6 + side * 5 + i] = 0u;  // zero cols 12..31
    hb[r * KPAD + 12] = one;  // bias column (dup same-addr/same-data writes OK)
  }
  lds_fence();

  // ---- layer 1: one K-tile, 20 N-tiles, relu -> LDS ----
  {
    v16bf a1 = load_a(hb, lane, 0);
    for (int nt = 0; nt < NT_H; ++nt) {
      v8f acc = {};  // C = inline 0 (bias comes from the K=12 row)
      acc = __builtin_amdgcn_wmma_f32_16x16x32_bf16(
          false, a1, false, load_b(w1p, nt, lane), (short)0, acc, false, false);
#pragma unroll
      for (int j = 0; j < 8; ++j)
        hb[(mbase + j) * KPAD + nt * 16 + ncol] = (__bf16)fmaxf(acc[j], 0.0f);
    }
    hb[ncol * KPAD + 300] = one;  // bias column for layer 2
  }
  lds_fence();

  // ---- cache all A tiles of h1, then layer 2 (200 WMMAs) ----
  v16bf A[KT_H];
#pragma unroll
  for (int kt = 0; kt < KT_H; ++kt) A[kt] = load_a(hb, lane, kt);
  lds_fence();  // reads done before overwriting hb with h2

  for (int nt = 0; nt < NT_H; ++nt) {
    v8f acc = {};
#pragma unroll
    for (int kt = 0; kt < KT_H; ++kt)
      acc = __builtin_amdgcn_wmma_f32_16x16x32_bf16(
          false, A[kt], false, load_b(w2p, kt * NT_H + nt, lane), (short)0, acc,
          false, false);
#pragma unroll
    for (int j = 0; j < 8; ++j)
      hb[(mbase + j) * KPAD + nt * 16 + ncol] = (__bf16)fmaxf(acc[j], 0.0f);
  }
  hb[ncol * KPAD + 300] = one;  // bias column for layer 3
  lds_fence();

  // ---- layer 3 (130 WMMAs) fused with KL reduction ----
#pragma unroll
  for (int kt = 0; kt < KT_H; ++kt) A[kt] = load_a(hb, lane, kt);

  float kl = 0.0f;
  for (int nt = 0; nt < NT_O; ++nt) {
    const int n = nt * 16 + ncol;
    v8f acc = {};
#pragma unroll
    for (int kt = 0; kt < KT_H; ++kt)
      acc = __builtin_amdgcn_wmma_f32_16x16x32_bf16(
          false, A[kt], false, load_b(w3p, kt * NT_O + nt, lane), (short)0, acc,
          false, false);
    const bool is_mu = (n < MSG_DIM);  // cols >=200 are exact zeros -> contribute 0
#pragma unroll
    for (int j = 0; j < 8; ++j) {
      float v = acc[j];
      float t = is_mu ? (0.5f * v * v) : (0.5f * (__expf(v) - v - 1.0f));
      kl += t;
    }
  }
  atomicAdd(&klsum, kl);
  __syncthreads();
  if (tid == 0) atomicAdd(out, klsum * (1.0f / (float)N_EDGES));
}

// ---------------- launch ----------------
extern "C" void kernel_launch(void* const* d_in, const int* in_sizes, int n_in,
                              void* d_out, int out_size, void* d_ws, size_t ws_size,
                              hipStream_t stream) {
  const float* x   = (const float*)d_in[0];
  const int*   ei  = (const int*)d_in[1];
  const float* y   = (const float*)d_in[2];
  const float* tgt = (const float*)d_in[3];
  const float* W1  = (const float*)d_in[4];
  const float* b1  = (const float*)d_in[5];
  const float* W2  = (const float*)d_in[6];
  const float* b2  = (const float*)d_in[7];
  const float* W3  = (const float*)d_in[8];
  const float* b3  = (const float*)d_in[9];

  char* ws = (char*)d_ws;
  __bf16* w1p = (__bf16*)(ws + W1P_OFF);
  __bf16* w2p = (__bf16*)(ws + W2P_OFF);
  __bf16* w3p = (__bf16*)(ws + W3P_OFF);
  float*  out = (float*)d_out;

  zero_out_kernel<<<1, 1, 0, stream>>>(out);

  const int prep_n = W1P_CNT + W2P_CNT + W3P_CNT;
  prep_kernel<<<(prep_n + 255) / 256, 256, 0, stream>>>(
      W1, b1, W2, b2, W3, b3, w1p, w2p, w3p);

  base_loss_kernel<<<(N_NODES * 2 + 255) / 256, 256, 0, stream>>>(y, tgt, out);

  const int blocks = N_EDGES / (16 * WAVES);  // 15625, exact
  fused_edge_mlp<<<blocks, 32 * WAVES, 0, stream>>>(x, ei, w1p, w2p, w3p, out);
}
